// LIF_SNN_17575006175870
// MI455X (gfx1250) — matrix operand order repaired and checked by reading, not compiled, not run on hardware
//
#include <hip/hip_runtime.h>

#define BB 256
#define TT 1000
#define NN 128
#define BN (BB * NN)   // 32768

typedef __attribute__((ext_vector_type(16))) _Float16 v16h;
typedef __attribute__((ext_vector_type(8)))  float    v8f;

// ---------------------------------------------------------------------------
// Pass 1: serial-T LIF recurrence. thread = (b, n); 1024 wave32s, no sync.
// Streams spikes + (v,i,s) states with non-temporal b32 stores (write-only
// 525 MB >> 192 MB L2). The l2 readout is deferred to pass 2 so the serial
// loop has zero cross-lane communication.
// ---------------------------------------------------------------------------
__global__ __launch_bounds__(256) void snn_pass1(
    const float* __restrict__ x,      // [B, T]  (input_batch [B,T,1])
    const float* __restrict__ st0,    // [3, B, N]
    const float* __restrict__ w1,     // [N]
    const float* __restrict__ bias,   // [N]
    const float* __restrict__ lki,    // [N]
    const float* __restrict__ lkv,    // [N]
    const float* __restrict__ thr,    // [N]
    float* __restrict__ spikes,       // [T, B, N]
    float* __restrict__ states)       // [T, 3, B, N]
{
    const int g = blockIdx.x * 256 + threadIdx.x;   // 0..32767
    const int b = g >> 7;
    const int n = g & 127;

    float v = st0[0 * BN + b * NN + n];
    float i = st0[1 * BN + b * NN + n];
    float s = st0[2 * BN + b * NN + n];

    const float W  = w1[n];
    const float Bi = bias[n];
    const float Li = lki[n];
    const float Lv = lkv[n];
    const float Th = thr[n];

    const float* __restrict__ xp = x + b * TT;      // wave-uniform address
    float* sp = spikes + (size_t)b * NN + n;
    float* vp = states + (size_t)b * NN + n;

    #pragma unroll 4
    for (int t = 0; t < TT; ++t) {
        const float xt = xp[t];
        // i_new = leak_i*i + x*w1 + bias
        i = __builtin_fmaf(Li, i, __builtin_fmaf(xt, W, Bi));
        // v_new = leak_v*v*(1-s) + i_new
        v = __builtin_fmaf(Lv * v, 1.0f - s, i);
        // s_new = (v_new - thres) > 0
        s = (v > Th) ? 1.0f : 0.0f;

        __builtin_nontemporal_store(s, sp);           // spikes[t,b,n]
        __builtin_nontemporal_store(v, vp);           // states[t,0,b,n]
        __builtin_nontemporal_store(i, vp + BN);      // states[t,1,b,n]
        __builtin_nontemporal_store(s, vp + 2 * BN);  // states[t,2,b,n]

        sp += BN;
        vp += 3 * BN;
    }
}

// ---------------------------------------------------------------------------
// Pass 2: decoded raw dots d[r] = spikes[r,:] . w2  for r = t*B+b (256000 rows)
// via v_wmma_f32_16x16x32_f16. One wave per 16-row tile (rows contiguous in
// memory: 16 consecutive b of one t). A = spikes as f16 (exact 0/1) in the
// documented 16-bit A 16x32 layout; B = w2 replicated across all 16 columns,
// split into hi+lo f16 halves (two WMMAs per chunk) for f32-level accuracy.
// ---------------------------------------------------------------------------
__global__ __launch_bounds__(256) void snn_pass2(
    const float* __restrict__ spikes,   // [T*B, N] f32 (values 0/1)
    const float* __restrict__ w2,       // [N]
    float* __restrict__ dec)            // [T*B] raw dots out
{
    const int tile = (blockIdx.x * 256 + threadIdx.x) >> 5;  // 16000 tiles
    const int lane = threadIdx.x & 31;
    const int m    = lane & 15;          // A row owned by this lane
    const int ksel = lane >> 4;          // 0: K{0..7,16..23}; 1: K{8..15,24..31}
    const int R    = tile * 16;          // first row of this tile

    // B fragments: every column = w2 chunk; hi/lo split so hi+lo == f32 w2.
    v16h bhi[4], blo[4];
    #pragma unroll
    for (int c = 0; c < 4; ++c) {
        #pragma unroll
        for (int e = 0; e < 16; ++e) {
            const float wf = w2[c * 32 + ((e >> 3) << 4) + ksel * 8 + (e & 7)];
            const _Float16 h = (_Float16)wf;
            bhi[c][e] = h;
            blo[c][e] = (_Float16)(wf - (float)h);
        }
    }

    const float* __restrict__ rp = spikes + (size_t)(R + m) * NN;
    v8f acc = {};
    #pragma unroll
    for (int c = 0; c < 4; ++c) {
        const int g0 = c * 32 + ksel * 8;        // VGPRs 0-3: 8 contiguous K
        const int g1 = c * 32 + 16 + ksel * 8;   // VGPRs 4-7: 8 contiguous K
        v16h a;
        #pragma unroll
        for (int e = 0; e < 8; ++e)
            a[e]     = (_Float16)__builtin_nontemporal_load(rp + g0 + e);
        #pragma unroll
        for (int e = 0; e < 8; ++e)
            a[8 + e] = (_Float16)__builtin_nontemporal_load(rp + g1 + e);

        acc = __builtin_amdgcn_wmma_f32_16x16x32_f16(
            false, a, false, bhi[c], (short)0, acc, false, false);
        acc = __builtin_amdgcn_wmma_f32_16x16x32_f16(
            false, a, false, blo[c], (short)0, acc, false, false);
    }

    // D[m, n] == dot(row m) for every n.  C/D layout: VGPR r = {M=r lanes0-15,
    // M=8+r lanes16-31}. So lane 0 holds dots 0..7, lane 16 holds dots 8..15.
    if (lane == 0) {
        *reinterpret_cast<float4*>(dec + R)     = make_float4(acc[0], acc[1], acc[2], acc[3]);
        *reinterpret_cast<float4*>(dec + R + 4) = make_float4(acc[4], acc[5], acc[6], acc[7]);
    } else if (lane == 16) {
        *reinterpret_cast<float4*>(dec + R + 8)  = make_float4(acc[0], acc[1], acc[2], acc[3]);
        *reinterpret_cast<float4*>(dec + R + 12) = make_float4(acc[4], acc[5], acc[6], acc[7]);
    }
}

// ---------------------------------------------------------------------------
// Pass 3: in-place linear scan over T for the leaky-integrator readout:
// li[t] = l2_leak * li[t-1] + d[t].  256 threads (one per b), coalesced.
// ---------------------------------------------------------------------------
__global__ __launch_bounds__(256) void snn_pass3(
    const float* __restrict__ li0,    // [B]
    const float* __restrict__ leak,   // [1]
    float* __restrict__ dec)          // [T, B] in: raw dots, out: decoded
{
    const int b = threadIdx.x;
    float li = li0[b];
    const float L = leak[0];
    float* p = dec + b;
    #pragma unroll 4
    for (int t = 0; t < TT; ++t) {
        const float d = p[0];
        li = __builtin_fmaf(L, li, d);
        p[0] = li;
        p += BB;
    }
}

extern "C" void kernel_launch(void* const* d_in, const int* in_sizes, int n_in,
                              void* d_out, int out_size, void* d_ws, size_t ws_size,
                              hipStream_t stream) {
    (void)in_sizes; (void)n_in; (void)out_size; (void)d_ws; (void)ws_size;

    const float* x    = (const float*)d_in[0];  // input_batch [B,T,1]
    const float* st0  = (const float*)d_in[1];  // state_snn   [3,B,N]
    const float* li0  = (const float*)d_in[2];  // state_LI    [B,1]
    const float* w1   = (const float*)d_in[3];  // l1_weights  [N,1]
    const float* bias = (const float*)d_in[4];  // l1_bias     [N]
    const float* lki  = (const float*)d_in[5];  // l1_leak_i   [N]
    const float* lkv  = (const float*)d_in[6];  // l1_leak_v   [N]
    const float* thr  = (const float*)d_in[7];  // l1_thres    [N]
    const float* w2   = (const float*)d_in[8];  // l2_weights  [1,N]
    const float* lk2  = (const float*)d_in[9];  // l2_leak     [1]

    float* out    = (float*)d_out;
    float* spikes = out;                          // [T,B,N]
    float* states = out + (size_t)TT * BN;        // [T,3,B,N]
    float* dec    = out + (size_t)TT * BN * 4;    // [T,B]

    // Pass 1: 32768 threads = 1024 wave32s, serial T loop.
    snn_pass1<<<dim3(BN / 256), dim3(256), 0, stream>>>(
        x, st0, w1, bias, lki, lkv, thr, spikes, states);

    // Pass 2: T*B/16 = 16000 tiles, 8 waves per 256-thread block.
    snn_pass2<<<dim3((TT * BB / 16) / 8), dim3(256), 0, stream>>>(spikes, w2, dec);

    // Pass 3: one 256-thread block scans T.
    snn_pass3<<<dim3(1), dim3(BB), 0, stream>>>(li0, lk2, dec);
}
